// GAT_50036368998546
// MI455X (gfx1250) — matrix operand order repaired and checked by reading, not compiled, not run on hardware
//
#include <hip/hip_runtime.h>
#include <hip/hip_bf16.h>

// ---------------------------------------------------------------------------
// Problem constants (from the reference)
// ---------------------------------------------------------------------------
constexpr int N_NODES = 512;
constexpr int D_IN    = 128;
constexpr int E_EDGES = 32768;
constexpr int H_HEADS = 2;
constexpr int C_CH    = 200;
constexpr int HC      = H_HEADS * C_CH;     // 400
constexpr int ETOT    = E_EDGES + N_NODES;  // self loops appended: 33280
constexpr int NHC     = N_NODES * HC;       // 204800
constexpr int FC_IN   = NHC;                // 204800
constexpr int FC_OUT  = 1000;
constexpr float LEAKY = 0.2f;

constexpr int KP0 = 128;                    // layer-0 K (already 32-aligned)
constexpr int KP1 = 416;                    // layers 1/2: 400 padded to 32

typedef __attribute__((ext_vector_type(16))) __bf16 v16bf;
typedef __attribute__((ext_vector_type(8)))  float  v8f;

// float -> bf16 (round-to-nearest-even), pure bit ops so it lowers inline
static __device__ __forceinline__ __bf16 f2bf(float f) {
    union { float f; unsigned u; } v; v.f = f;
    unsigned r = v.u + 0x7FFFu + ((v.u >> 16) & 1u);
    union { unsigned short s; __bf16 b; } o;
    o.s = (unsigned short)(r >> 16);
    return o.b;
}

static __device__ __forceinline__ void atomicMaxF(float* addr, float val) {
    unsigned int* ua = (unsigned int*)addr;
    unsigned int old = *ua;
    while (true) {
        float cur = __uint_as_float(old);
        if (cur >= val) break;
        unsigned int assumed = old;
        old = atomicCAS(ua, assumed, __float_as_uint(val));
        if (old == assumed) break;
    }
}

// ---------------------------------------------------------------------------
// Pack fp32 activations X[M,K] -> bf16 Xb[M,KP], zero-padding K..KP-1.
// ---------------------------------------------------------------------------
__global__ void pack_x_bf16(const float* __restrict__ X, __bf16* __restrict__ Xb,
                            int M, int K, int KPad) {
    const int idx = blockIdx.x * blockDim.x + threadIdx.x;
    if (idx >= M * KPad) return;
    const int r = idx / KPad, k = idx % KPad;
    Xb[idx] = (k < K) ? f2bf(X[r * K + k]) : f2bf(0.0f);
}

// ---------------------------------------------------------------------------
// Pack + transpose fp32 W[K,Nn] -> bf16 Wt[Nn,KP], zero-padding K..KP-1.
// Makes the WMMA B-fragment (a column of W) contiguous in memory.
// ---------------------------------------------------------------------------
__global__ void pack_wt_bf16(const float* __restrict__ W, __bf16* __restrict__ Wt,
                             int K, int Nn, int KPad) {
    const int idx = blockIdx.x * blockDim.x + threadIdx.x;
    if (idx >= Nn * KPad) return;
    const int n = idx / KPad, k = idx % KPad;
    Wt[idx] = (k < K) ? f2bf(W[k * Nn + n]) : f2bf(0.0f);
}

// ---------------------------------------------------------------------------
// Y[M,Nn] = Xb[M,KP](bf16) @ Wt[Nn,KP]^T(bf16) + bias, fp32 accumulate.
// One wave per 16x16 tile, 4 waves/block. KP compile-time -> full unroll.
//
// CDNA5 fragment layouts (ISA 7.12.2), all contiguous 16-byte loads:
//   A (16x32 bf16): lane l15 = row; a[0..7] at K=k0+half*8, a[8..15] at
//                   K=k0+16+half*8 -> two global_load_b128 per step.
//   B (32x16 bf16): lane l15 = column; b[0..7] at K=k0+half*8, b[8..15] at
//                   K=k0+16+half*8 in the transposed Wt -> two global_load_b128 per step.
//   C/D (16x16 f32): VGPR r -> row half*8+r, lane l15 -> column.
// No guards: padding guarantees every chunk is full, EXEC stays all-ones.
// ---------------------------------------------------------------------------
template<int KPAD>
__global__ void gemm_bias_wmma(const __bf16* __restrict__ Xb,
                               const __bf16* __restrict__ Wt,
                               const float* __restrict__ bias,
                               float* __restrict__ Y,
                               int M, int Nn) {
    const int lane = threadIdx.x & 31;
    const int wave = threadIdx.x >> 5;
    const int nTilesN = Nn >> 4;                 // Nn multiple of 16
    const int nTiles  = (M >> 4) * nTilesN;
    const int tile = blockIdx.x * 4 + wave;
    if (tile >= nTiles) return;                  // wave-uniform exit

    const int tm = (tile / nTilesN) << 4;
    const int tn = (tile % nTilesN) << 4;
    const int half = lane >> 4;                  // 0: lanes 0-15, 1: lanes 16-31
    const int l15  = lane & 15;

    const uint4* __restrict__ pa =
        reinterpret_cast<const uint4*>(Xb + (size_t)(tm + l15) * KPAD);
    const uint4* __restrict__ pb =
        reinterpret_cast<const uint4*>(Wt + (size_t)(tn + l15) * KPAD);

    v8f acc = {};
    #pragma unroll
    for (int k0 = 0; k0 < KPAD; k0 += 32) {
        const int q = k0 >> 3;                   // uint4 (8 x bf16) index
        union { v16bf v; uint4 u[2]; } af, bfrag;
        af.u[0]    = pa[q + half];               // K = k0 + half*8 .. +7
        af.u[1]    = pa[q + 2 + half];           // K = k0+16+half*8 .. +7
        bfrag.u[0] = pb[q + half];               // K = k0 + half*8 .. +7
        bfrag.u[1] = pb[q + 2 + half];           // K = k0+16+half*8 .. +7
        acc = __builtin_amdgcn_wmma_f32_16x16x32_bf16(
                  false, af.v, false, bfrag.v, (short)0, acc, false, false);
    }

    const int bcol = tn + l15;
    const float bv = bias[bcol];
    #pragma unroll
    for (int r = 0; r < 8; ++r) {
        const int row = tm + half * 8 + r;
        Y[row * Nn + bcol] = acc[r] + bv;
    }
}

// ---------------------------------------------------------------------------
// Per-layer init: zero aggregation accumulator, softmax denom; m = -inf
// ---------------------------------------------------------------------------
__global__ void init_layer_kernel(float* __restrict__ acc,
                                  float* __restrict__ mmax,
                                  float* __restrict__ denom) {
    const int idx = blockIdx.x * blockDim.x + threadIdx.x;
    if (idx < NHC) acc[idx] = 0.0f;
    if (idx < N_NODES * H_HEADS) {
        mmax[idx]  = -__builtin_inff();
        denom[idx] = 0.0f;
    }
}

// ---------------------------------------------------------------------------
// One wave per edge: logits[e,h] = att[h,:] . leaky_relu(xl[src]+xr[dst]);
// segment max via atomicMaxF on dst. wave32 xor-shuffle reduction.
// ---------------------------------------------------------------------------
__global__ void edge_logits_kernel(const float* __restrict__ xl,
                                   const float* __restrict__ xr,
                                   const int* __restrict__ ei,
                                   const float* __restrict__ att,
                                   float* __restrict__ logits,
                                   float* __restrict__ mmax) {
    const int lane = threadIdx.x & 31;
    const int e = (blockIdx.x * blockDim.x + threadIdx.x) >> 5;
    if (e >= ETOT) return;
    const int src = (e < E_EDGES) ? ei[e]           : (e - E_EDGES);
    const int dst = (e < E_EDGES) ? ei[E_EDGES + e] : (e - E_EDGES);
    const float* pl = xl + src * HC;
    const float* pr = xr + dst * HC;
    #pragma unroll
    for (int h = 0; h < H_HEADS; ++h) {
        float s = 0.0f;
        for (int c = lane; c < C_CH; c += 32) {
            float v = pl[h * C_CH + c] + pr[h * C_CH + c];
            v = (v > 0.0f) ? v : (LEAKY * v);
            s += v * att[h * C_CH + c];
        }
        s += __shfl_xor(s, 16);
        s += __shfl_xor(s, 8);
        s += __shfl_xor(s, 4);
        s += __shfl_xor(s, 2);
        s += __shfl_xor(s, 1);
        if (lane == 0) {
            logits[e * H_HEADS + h] = s;
            atomicMaxF(&mmax[dst * H_HEADS + h], s);
        }
    }
}

// ---------------------------------------------------------------------------
// ex = exp(logit - m[dst]); denom[dst] += ex
// ---------------------------------------------------------------------------
__global__ void edge_exp_kernel(const float* __restrict__ logits,
                                const float* __restrict__ mmax,
                                const int* __restrict__ ei,
                                float* __restrict__ ex,
                                float* __restrict__ denom) {
    const int idx = blockIdx.x * blockDim.x + threadIdx.x;
    if (idx >= ETOT * H_HEADS) return;
    const int e = idx / H_HEADS;
    const int h = idx % H_HEADS;
    const int dst = (e < E_EDGES) ? ei[E_EDGES + e] : (e - E_EDGES);
    const float v = __expf(logits[idx] - mmax[dst * H_HEADS + h]);
    ex[idx] = v;
    atomicAdd(&denom[dst * H_HEADS + h], v);
}

// ---------------------------------------------------------------------------
// acc[dst, :] += (ex/denom[dst]) * xl[src, :]  -- thread per (edge, channel).
// acc is 800 KB -> L2 resident atomics.
// ---------------------------------------------------------------------------
__global__ void edge_aggregate_kernel(const float* __restrict__ ex,
                                      const float* __restrict__ denom,
                                      const float* __restrict__ xl,
                                      const int* __restrict__ ei,
                                      float* __restrict__ acc) {
    const int idx = blockIdx.x * blockDim.x + threadIdx.x;
    if (idx >= ETOT * HC) return;
    const int e = idx / HC;
    const int r = idx % HC;
    const int h = r / C_CH;
    const int src = (e < E_EDGES) ? ei[e]           : (e - E_EDGES);
    const int dst = (e < E_EDGES) ? ei[E_EDGES + e] : (e - E_EDGES);
    const float alpha = ex[e * H_HEADS + h] / denom[dst * H_HEADS + h];
    atomicAdd(&acc[dst * HC + r], alpha * xl[src * HC + r]);
}

// ---------------------------------------------------------------------------
// h_out = (acc + bias) [+ relu]
// ---------------------------------------------------------------------------
__global__ void bias_act_kernel(const float* __restrict__ acc,
                                const float* __restrict__ b,
                                float* __restrict__ hout,
                                int do_relu) {
    const int idx = blockIdx.x * blockDim.x + threadIdx.x;
    if (idx >= NHC) return;
    float v = acc[idx] + b[idx % HC];
    if (do_relu) v = fmaxf(v, 0.0f);
    hout[idx] = v;
}

// ---------------------------------------------------------------------------
// FC: bandwidth-bound matvec (819 MB weight stream), split-K + atomic combine.
// Thread j (coalesced over FC_OUT) streams its column over an i-chunk.
// ---------------------------------------------------------------------------
constexpr int FC_ICHUNK = 256;

__global__ void fc_zero_kernel(float* __restrict__ part) {
    const int i = blockIdx.x * blockDim.x + threadIdx.x;
    if (i < FC_OUT) part[i] = 0.0f;
}

__global__ void fc_partial_kernel(const float* __restrict__ h,
                                  const float* __restrict__ fcW,
                                  float* __restrict__ part) {
    const int j  = blockIdx.x * blockDim.x + threadIdx.x;
    const int i0 = blockIdx.y * FC_ICHUNK;
    if (j >= FC_OUT) return;
    float s = 0.0f;
    #pragma unroll 4
    for (int i = i0; i < i0 + FC_ICHUNK; ++i) {
        __builtin_prefetch(&fcW[(i + 16) * FC_OUT + j], 0, 0);
        s = fmaf(h[i], fcW[i * FC_OUT + j], s);
    }
    atomicAdd(&part[j], s);
}

__global__ void fc_epilogue_kernel(const float* __restrict__ part,
                                   const float* __restrict__ fcb,
                                   float* __restrict__ out) {
    const int j = blockIdx.x * blockDim.x + threadIdx.x;
    if (j >= FC_OUT) return;
    out[j] = fmaxf(part[j] + fcb[j], 0.0f);
}

// ---------------------------------------------------------------------------
// Host orchestration
// ---------------------------------------------------------------------------
extern "C" void kernel_launch(void* const* d_in, const int* in_sizes, int n_in,
                              void* d_out, int out_size, void* d_ws, size_t ws_size,
                              hipStream_t stream) {
    (void)in_sizes; (void)n_in; (void)out_size; (void)ws_size;

    const float* x   = (const float*)d_in[0];
    const int*   ei  = (const int*)d_in[1];
    const float* Wl[3]  = {(const float*)d_in[2],  (const float*)d_in[8],  (const float*)d_in[14]};
    const float* bl[3]  = {(const float*)d_in[3],  (const float*)d_in[9],  (const float*)d_in[15]};
    const float* Wr[3]  = {(const float*)d_in[4],  (const float*)d_in[10], (const float*)d_in[16]};
    const float* br[3]  = {(const float*)d_in[5],  (const float*)d_in[11], (const float*)d_in[17]};
    const float* att[3] = {(const float*)d_in[6],  (const float*)d_in[12], (const float*)d_in[18]};
    const float* bb[3]  = {(const float*)d_in[7],  (const float*)d_in[13], (const float*)d_in[19]};
    const float* fcW = (const float*)d_in[20];
    const float* fcb = (const float*)d_in[21];
    float* out = (float*)d_out;

    // workspace carve-out: fp32 region then bf16 region (~5 MB total).
    // Every sub-buffer size is a multiple of 16 bytes -> b128 alignment holds.
    float* ws     = (float*)d_ws;
    float* hbuf   = ws; ws += NHC;              // layer activations (fp32)
    float* xl     = ws; ws += NHC;
    float* xr     = ws; ws += NHC;
    float* acc    = ws; ws += NHC;              // segment-sum accumulator
    float* logits = ws; ws += ETOT * H_HEADS;
    float* exb    = ws; ws += ETOT * H_HEADS;
    float* mmax   = ws; ws += 1024;
    float* denom  = ws; ws += 1024;
    float* part   = ws; ws += 1024;             // FC split-K partials
    __bf16* bws   = (__bf16*)ws;
    __bf16* xb    = bws; bws += N_NODES * KP1;  // padded bf16 activations
    __bf16* wlt   = bws; bws += HC * KP1;       // packed/transposed Wl
    __bf16* wrt   = bws; bws += HC * KP1;       // packed/transposed Wr

    const int gemmTiles  = (N_NODES / 16) * (HC / 16);   // 32 * 25 = 800
    const int gemmBlocks = (gemmTiles + 3) / 4;          // 4 waves / block

    for (int l = 0; l < 3; ++l) {
        const int K  = (l == 0) ? D_IN : HC;
        const int KPad = (l == 0) ? KP0 : KP1;
        const float* hin = (l == 0) ? x : hbuf;

        // bf16 pack: activations (padded) + both weight matrices (transposed)
        pack_x_bf16<<<(N_NODES * KPad + 255) / 256, 256, 0, stream>>>(hin, xb, N_NODES, K, KPad);
        pack_wt_bf16<<<(HC * KPad + 255) / 256, 256, 0, stream>>>(Wl[l], wlt, K, HC, KPad);
        pack_wt_bf16<<<(HC * KPad + 255) / 256, 256, 0, stream>>>(Wr[l], wrt, K, HC, KPad);

        if (l == 0) {
            gemm_bias_wmma<KP0><<<gemmBlocks, 128, 0, stream>>>(xb, wlt, bl[l], xl, N_NODES, HC);
            gemm_bias_wmma<KP0><<<gemmBlocks, 128, 0, stream>>>(xb, wrt, br[l], xr, N_NODES, HC);
        } else {
            gemm_bias_wmma<KP1><<<gemmBlocks, 128, 0, stream>>>(xb, wlt, bl[l], xl, N_NODES, HC);
            gemm_bias_wmma<KP1><<<gemmBlocks, 128, 0, stream>>>(xb, wrt, br[l], xr, N_NODES, HC);
        }

        init_layer_kernel<<<(NHC + 255) / 256, 256, 0, stream>>>(acc, mmax, denom);

        edge_logits_kernel<<<(ETOT + 7) / 8, 256, 0, stream>>>(xl, xr, ei, att[l], logits, mmax);
        edge_exp_kernel<<<(ETOT * H_HEADS + 255) / 256, 256, 0, stream>>>(logits, mmax, ei, exb, denom);
        edge_aggregate_kernel<<<(ETOT * HC + 255) / 256, 256, 0, stream>>>(exb, denom, xl, ei, acc);

        bias_act_kernel<<<(NHC + 255) / 256, 256, 0, stream>>>(acc, bb[l], hbuf, (l != 2) ? 1 : 0);
    }

    fc_zero_kernel<<<(FC_OUT + 255) / 256, 256, 0, stream>>>(part);
    dim3 fcGrid((FC_OUT + 255) / 256, FC_IN / FC_ICHUNK);
    fc_partial_kernel<<<fcGrid, 256, 0, stream>>>(hbuf, fcW, part);
    fc_epilogue_kernel<<<(FC_OUT + 255) / 256, 256, 0, stream>>>(part, fcb, out);
}
